// SlidingWindowAttention_70093866270824
// MI455X (gfx1250) — matrix operand order repaired and checked by reading, not compiled, not run on hardware
//
#include <hip/hip_runtime.h>

// Shapes from the reference
#define BB   2
#define SS   1024
#define DD   65      // head dim == 2W+1
#define HH   8
#define C3D  195     // 3*D (x row)
#define HD   520     // H*D
#define QKC  1040    // q+k columns of w_qkv (v is dead)
#define KEXT 544     // K padded to multiple of 32 for 16x16x32 WMMA
#define PKS  552     // LDS row stride (elements) -> 1104B, 16B aligned, bank-friendly
#define PM   80      // M/N rows padded to multiple of 16

typedef _Float16 v16h __attribute__((ext_vector_type(16)));
typedef _Float16 v8h  __attribute__((ext_vector_type(8)));
typedef float    v8f  __attribute__((ext_vector_type(8)));
typedef unsigned int u32x4 __attribute__((ext_vector_type(4)));
typedef int          i32x4 __attribute__((ext_vector_type(4)));
typedef int          i32x8 __attribute__((ext_vector_type(8)));

// ---------------- Kernel 1: qk = x @ w_qkv[:1040].T + b ----------------
__global__ void qk_gemm_kernel(const float* __restrict__ x,
                               const float* __restrict__ w_qkv,
                               const float* __restrict__ b_qkv,
                               float* __restrict__ qk) {
  __shared__ float xs[32][C3D];
  const int bs0 = blockIdx.x * 32;
  for (int idx = threadIdx.x; idx < 32 * C3D; idx += 256) {
    int r = idx / C3D, i = idx % C3D;
    xs[r][i] = x[(bs0 + r) * C3D + i];
  }
  __syncthreads();
  for (int c = threadIdx.x; c < QKC; c += 256) {
    float acc[32];
    const float bias = b_qkv[c];
#pragma unroll
    for (int r = 0; r < 32; ++r) acc[r] = bias;
    const float* wr = w_qkv + c * C3D;
    for (int i = 0; i < C3D; ++i) {
      float wv = wr[i];
#pragma unroll
      for (int r = 0; r < 32; ++r) acc[r] = fmaf(xs[r][i], wv, acc[r]);
    }
#pragma unroll
    for (int r = 0; r < 32; ++r) qk[(bs0 + r) * QKC + c] = acc[r];
  }
}

// ---------------- Kernel 2: w_proj -> padded f16 [PM][PKS] ----------------
__global__ void wp_prep_kernel(const float* __restrict__ w_proj,
                               _Float16* __restrict__ wp16) {
  int idx = blockIdx.x * 256 + threadIdx.x;
  if (idx >= PM * PKS) return;
  int n = idx / PKS, k = idx % PKS;
  float v = (n < DD && k < HD) ? w_proj[n * HD + k] : 0.0f;
  wp16[idx] = (_Float16)v;
}

// ---------------- Kernel 3: fused outer-product / softmax / WMMA GEMM ----------------
// One workgroup per (b,s). LDS: A probs (f16, 80x552), Wp staged by the TDM
// (f16, 80x552), permuted rel table (f32 65x65), q/k rows. ~197.7 KB < 320 KB/WGP.
__global__ void __launch_bounds__(256)
fused_attn_kernel(const float* __restrict__ qk,
                  const _Float16* __restrict__ wp16,
                  const float* __restrict__ rel_table,
                  const float* __restrict__ b_proj,
                  float* __restrict__ out) {
  extern __shared__ char smem[];
  _Float16* As    = (_Float16*)smem;                       // PM*PKS f16 = 88320 B
  _Float16* Bs    = (_Float16*)(smem + PM * PKS * 2);      // PM*PKS f16 = 88320 B
  float*    rel_s = (float*)(smem + 2 * PM * PKS * 2);     // 65*65 f32  = 16900 B
  float*    qs    = (float*)(smem + 2 * PM * PKS * 2 + DD * DD * 4);  // 2080 B
  float*    ks    = qs + HD;                                          // 2080 B

  const int tid  = threadIdx.x;
  const int bsid = blockIdx.x;          // [0, B*S)
  const int b    = bsid / SS;
  const int s    = bsid % SS;

  // ---- Kick off the TDM DMA of Wp(f16, padded) into Bs; runs concurrently
  // ---- with the whole softmax phase. One descriptor, issued by wave 0.
  if (tid < 32) {
    const unsigned long long ga = (unsigned long long)wp16;
    const unsigned lds_a = (unsigned)(unsigned long long)Bs; // LDS aperture: addr[31:0] is the LDS offset
    const unsigned n64 = (PM * PKS * 2) / 8;                 // 11040 8-byte elements

    u32x4 g0;
    g0[0] = 1u;                                              // count=1 (valid user D#)
    g0[1] = lds_a;                                           // lds_addr
    g0[2] = (unsigned)ga;                                    // global_addr[31:0]
    g0[3] = (unsigned)((ga >> 32) & 0x1FFFFFFu) | (2u << 30);// global_addr[56:32], type=2

    i32x8 g1;
    g1[0] = (3 << 16);                 // data_size=8B; no multicast/pad/iterate
    g1[1] = (int)((n64 & 0xFFFFu) << 16);   // tensor_dim0[15:0]  @bits 63:48
    g1[2] = (1 << 16);                 // tensor_dim0[31:16]=0; tensor_dim1=1 @bits 95:80
    g1[3] = (int)((n64 & 0xFFFFu) << 16);   // tile_dim0 @bits 127:112
    g1[4] = 1;                         // tile_dim1=1; tile_dim2=0
    g1[5] = (int)n64;                  // tensor_dim0_stride[31:0]
    g1[6] = 0;
    g1[7] = 0;

    i32x4 gz4 = {0, 0, 0, 0};
    i32x8 gz8 = {0, 0, 0, 0, 0, 0, 0, 0};
    __builtin_amdgcn_tensor_load_to_lds(g0, g1, gz4, gz4, gz8, 0);
  }

  // Stage q/k rows for this (b,s)
  const float* qkrow = qk + bsid * QKC;
  for (int i = tid; i < HD; i += 256) { qs[i] = qkrow[i]; ks[i] = qkrow[HD + i]; }

  // Stage rel table with the torch negative-index permutation: idx[d] = (d-32) mod 65 = (d+33)%65
  for (int idx = tid; idx < DD * DD; idx += 256) {
    int d = idx / DD, dv = idx % DD;
    rel_s[idx] = rel_table[((d + 33) % DD) * DD + dv];
  }

  // Zero A so padding rows/cols contribute 0 to the WMMA GEMM
  {
    uint4* asv = (uint4*)As;
    uint4 z; z.x = z.y = z.z = z.w = 0u;
    for (int i = tid; i < PM * PKS * 2 / 16; i += 256) asv[i] = z;
  }
  __syncthreads();

  // Softmax over d for each of the 520 (h,dv) columns; write f16 probs into
  // As laid out as the 65x520 "run matrix": (h,d,dv) -> row g/8, col (g%8)*65+dv, g=h*65+d.
  for (int c = tid; c < HD; c += 256) {
    const int h = c / DD, dv = c % DD;
    const float kc = ks[c];
    const float* qh = qs + h * DD;
    float mx = -3.0e38f;
    for (int d = 0; d < DD; ++d)
      mx = fmaxf(mx, fmaf(qh[d], kc, rel_s[d * DD + dv]));
    float sum = 0.0f;
    for (int d = 0; d < DD; ++d) {
      float m = fmaf(qh[d], kc, rel_s[d * DD + dv]);
      float e = __expf(m - mx);
      sum += e;
      int g = h * DD + d;
      As[(g >> 3) * PKS + (g & 7) * DD + dv] = (_Float16)e;
    }
    const float rs = 1.0f / sum;
    for (int d = 0; d < DD; ++d) {
      int g = h * DD + d;
      int off = (g >> 3) * PKS + (g & 7) * DD + dv;
      As[off] = (_Float16)((float)As[off] * rs);
    }
  }
  __syncthreads();

  // Make the TDM result visible: issuing wave drains TENSORcnt, then all waves sync.
  if (tid < 32) __builtin_amdgcn_s_wait_tensorcnt(0);
  __syncthreads();

  // O(65x65) = A_r(65x520) @ Wp^T via v_wmma_f32_16x16x32_f16.
  const int wid  = tid >> 5;
  const int lane = tid & 31;
  const int lh   = lane >> 4;   // half of wave
  const int lm   = lane & 15;

  for (int t = wid; t < 25; t += 8) {
    const int mt = t / 5, nt = t % 5;
    const _Float16* arow = As + (mt * 16 + lm) * PKS;   // A: lane row M = lane%16
    const _Float16* brow = Bs + (nt * 16 + lm) * PKS;   // B: lane col N = lane%16
    v8f acc = {0.f, 0.f, 0.f, 0.f, 0.f, 0.f, 0.f, 0.f};
    for (int k0 = 0; k0 < KEXT; k0 += 32) {
      v16h a, bfr;
      // A 16x32 f16 layout: VGPR0-3 <- K[8h,8h+8), VGPR4-7 <- K[16+8h,16+8h+8)
      *(v8h*)&a           = *(const v8h*)(arow + k0 + 8 * lh);
      *(((v8h*)&a) + 1)   = *(const v8h*)(arow + k0 + 16 + 8 * lh);
      // B 32x16 f16 layout: VGPR0-3 <- K[8h,8h+8), VGPR4-7 <- K[16+8h,16+8h+8)
      *(v8h*)&bfr         = *(const v8h*)(brow + k0 + 8 * lh);
      *(((v8h*)&bfr) + 1) = *(const v8h*)(brow + k0 + 16 + 8 * lh);
      acc = __builtin_amdgcn_wmma_f32_16x16x32_f16(false, a, false, bfr,
                                                   (short)0, acc, false, false);
    }
    // C/D layout: VGPR i -> M = i + 8*half, N = lane%16. Scatter through the
    // (b,s,q) -> (m=b*65+r, s') reinterpretation: t = s*65+q, r=t/1024, s'=t%1024.
    const int j = nt * 16 + lm;
    if (j < DD) {
      const float bp = b_proj[j];
#pragma unroll
      for (int i = 0; i < 8; ++i) {
        const int qrow = mt * 16 + i + 8 * lh;
        if (qrow < DD) {
          const int tq = s * DD + qrow;
          const int rr = tq >> 10;          // /1024
          const int sp = tq & (SS - 1);     // %1024
          out[((b * DD + rr) * SS + sp) * DD + j] = acc[i] + bp;
        }
      }
    }
  }
}

// ---------------- Launch ----------------
extern "C" void kernel_launch(void* const* d_in, const int* in_sizes, int n_in,
                              void* d_out, int out_size, void* d_ws, size_t ws_size,
                              hipStream_t stream) {
  const float* x         = (const float*)d_in[0];
  const float* w_qkv     = (const float*)d_in[1];
  const float* b_qkv     = (const float*)d_in[2];
  const float* rel_table = (const float*)d_in[3];
  const float* w_proj    = (const float*)d_in[4];
  const float* b_proj    = (const float*)d_in[5];
  float* out = (float*)d_out;

  // Workspace layout: [ qk: B*S*1040 f32 | wp16: PM*PKS f16 ]
  const size_t qk_bytes = (size_t)BB * SS * QKC * sizeof(float);   // 8,519,680 B
  float*    qk_ws = (float*)d_ws;
  _Float16* wp16  = (_Float16*)((char*)d_ws + qk_bytes);           // 88,320 B

  qk_gemm_kernel<<<(BB * SS) / 32, 256, 0, stream>>>(x, w_qkv, b_qkv, qk_ws);
  wp_prep_kernel<<<(PM * PKS + 255) / 256, 256, 0, stream>>>(w_proj, wp16);

  const size_t smem_bytes = (size_t)(2 * PM * PKS * 2) + DD * DD * 4 + 2 * HD * 4; // 197,700 B
  fused_attn_kernel<<<BB * SS, 256, smem_bytes, stream>>>(qk_ws, wp16, rel_table, b_proj, out);
}
